// RetentionLikeWalker_45071386804736
// MI455X (gfx1250) — matrix operand order repaired and checked by reading, not compiled, not run on hardware
//
#include <hip/hip_runtime.h>
#include <math.h>

#define B_DIM 8
#define N_DIM 4096
#define D_DIM 1024
#define CH    128
#define NC    (N_DIM / CH)
#define BN    (B_DIM * N_DIM)

// LDS slab geometry (per buffer): W hi/lo as 4 q-planes of [1024 cols x 16B],
// then A hi/lo as 4 q-planes of [64 rows x 16B].
#define W_MAT_BYTES   65536              // 4 * 16384
#define W_ALL_BYTES   131072             // hi + lo
#define A_MAT_BYTES   4096               // 4 * 1024
#define BUF_BYTES     139264             // W_ALL + 2*A_MAT

typedef __attribute__((ext_vector_type(16))) __bf16       v16bf;
typedef __attribute__((ext_vector_type(8)))  float        v8f;
typedef __attribute__((ext_vector_type(4)))  unsigned int u32x4;

__device__ __forceinline__ unsigned short f2bf(float f) {
  unsigned u = __float_as_uint(f);
  u += 0x7FFFu + ((u >> 16) & 1u);           // round-to-nearest-even
  return (unsigned short)(u >> 16);
}
__device__ __forceinline__ float bf2f(unsigned short h) {
  return __uint_as_float(((unsigned)h) << 16);
}
__device__ __forceinline__ float sigmoidf_(float x) { return 1.0f / (1.0f + expf(-x)); }
__device__ __forceinline__ float powi_(float a, int e) {
  float p = 1.f, b = a;
  while (e) { if (e & 1) p *= b; b *= b; e >>= 1; }
  return p;
}

// async copy of one 16-byte chunk: global -> LDS (tracked by ASYNCcnt)
__device__ __forceinline__ void async_b128(unsigned ldsByteOff, const void* g) {
  asm volatile("global_load_async_to_lds_b128 %0, %1, off"
               :: "v"(ldsByteOff), "v"((unsigned long long)(uintptr_t)g)
               : "memory");
}
__device__ __forceinline__ void wait_async0() {
  asm volatile("s_wait_asynccnt 0x0" ::: "memory");
}

// ---------------- Kernel 1: chunk-local EMA scan, write bf16 hi/lo + chunk-end carry ----
__global__ void ema_local_scan(const float* __restrict__ Z, const float* __restrict__ decay,
                               unsigned short* __restrict__ Shi, unsigned short* __restrict__ Slo,
                               float* __restrict__ E) {
  const int d = blockIdx.x * blockDim.x + threadIdx.x;
  const int c = blockIdx.y, b = blockIdx.z;
  const float a = sigmoidf_(decay[0]);
  const size_t base = ((size_t)b * N_DIM + (size_t)c * CH) * D_DIM + d;
  float s = 0.f;
  for (int k = 0; k < CH; ++k) {
    const size_t idx = base + (size_t)k * D_DIM;
    s = fmaf(a, s, Z[idx]);
    const unsigned short hi = f2bf(s);
    Shi[idx] = hi;
    Slo[idx] = f2bf(s - bf2f(hi));
  }
  E[((size_t)b * NC + c) * D_DIM + d] = s;
}

// ---------------- Kernel 2: exclusive EMA scan over chunk carries (coeff a^CH) ----------
__global__ void ema_chunk_carry(const float* __restrict__ decay, const float* __restrict__ E,
                                float* __restrict__ C) {
  const int d = blockIdx.x * blockDim.x + threadIdx.x;
  const int b = blockIdx.y;
  const float a  = sigmoidf_(decay[0]);
  const float aL = powi_(a, CH);
  float carry = 0.f;
  for (int c = 0; c < NC; ++c) {
    const size_t idx = ((size_t)b * NC + c) * D_DIM + d;
    C[idx] = carry;
    carry = fmaf(aL, carry, E[idx]);
  }
}

// ---------------- Kernel 3: add a^(k+1) * carry into each element, re-split hi/lo -------
__global__ void ema_fixup(const float* __restrict__ decay, const float* __restrict__ C,
                          unsigned short* __restrict__ Shi, unsigned short* __restrict__ Slo) {
  const int d = blockIdx.x * blockDim.x + threadIdx.x;
  const int n = blockIdx.y, b = blockIdx.z;
  const int c = n / CH, k = n % CH;
  const float a = sigmoidf_(decay[0]);
  const float p = powi_(a, k + 1);                       // uniform per block
  const float carry = C[((size_t)b * NC + c) * D_DIM + d];
  const size_t idx = ((size_t)b * N_DIM + n) * D_DIM + d;
  const float s = bf2f(Shi[idx]) + bf2f(Slo[idx]) + p * carry;
  const unsigned short hi = f2bf(s);
  Shi[idx] = hi;
  Slo[idx] = f2bf(s - bf2f(hi));
}

// ---------------- Kernel 4: split W into bf16 hi/lo -------------------------------------
__global__ void w_split(const float* __restrict__ W, unsigned short* __restrict__ Whi,
                        unsigned short* __restrict__ Wlo) {
  const size_t i = (size_t)blockIdx.x * blockDim.x + threadIdx.x;
  const float w = W[i];
  const unsigned short hi = f2bf(w);
  Whi[i] = hi;
  Wlo[i] = f2bf(w - bf2f(hi));
}

// ---------------- Kernel 5: WMMA GEMM (split-bf16) + fused LayerNorm --------------------
// Block = 1024 threads (32 waves) computes rows [64*bx, 64*bx+64) x all 1024 cols.
// Operand slabs for each 32-wide k-step are staged into LDS with async b128 copies
// (double buffered); WMMA operands come from ds_load_b128.
__global__ __launch_bounds__(1024)
void gemm_ln(const unsigned short* __restrict__ Shi, const unsigned short* __restrict__ Slo,
             const unsigned short* __restrict__ Whi, const unsigned short* __restrict__ Wlo,
             const float* __restrict__ bias, const float* __restrict__ gamma,
             const float* __restrict__ beta, float* __restrict__ Out) {
  __shared__ __align__(16) unsigned char slab[2][BUF_BYTES];   // 272 KB of the 320 KB LDS

  const int tid  = threadIdx.x;
  const int wave = tid >> 5;
  const int lane = tid & 31;
  const int g    = lane >> 4;     // lane group (K-interleave half)
  const int l15  = lane & 15;
  const int mw   = wave >> 3;     // row-wave 0..3
  const int cw   = wave & 7;      // col-wave 0..7

  const int rowBase = blockIdx.x * 64 + mw * 16;
  const int colBase = cw * 128;
  const unsigned ldsBase = (unsigned)(uintptr_t)(&slab[0][0]);

  // Stage the k-slab [k0, k0+32) of W(hi,lo) and this block's 64 A rows (hi,lo).
  auto stage = [&](int kt, int buf) {
    const int k0 = kt * 32;
    const unsigned bufO = ldsBase + (unsigned)buf * BUF_BYTES;
    // W: 8192 chunks of 16B -> 8 per thread. Consecutive lanes take consecutive
    // q-chunks of the same column (64B-contiguous global segments per 4 lanes).
#pragma unroll
    for (int i = 0; i < 8; ++i) {
      const int id  = i * 1024 + tid;       // 0..8191
      const int q   = id & 3;
      const int col = (id >> 2) & 1023;
      const int mat = id >> 12;             // 0=hi 1=lo
      const unsigned short* gsrc = (mat ? Wlo : Whi) + (size_t)col * D_DIM + k0 + 8 * q;
      const unsigned lds = bufO + (unsigned)(mat * W_MAT_BYTES + q * 16384 + col * 16);
      async_b128(lds, gsrc);
    }
    // A: 512 chunks -> threads 0..511 issue one each.
    if (tid < 512) {
      const int q   = tid & 3;
      const int row = (tid >> 2) & 63;
      const int mat = tid >> 8;
      const unsigned short* gsrc = (mat ? Slo : Shi)
          + (size_t)(blockIdx.x * 64 + row) * D_DIM + k0 + 8 * q;
      const unsigned lds = bufO + (unsigned)(W_ALL_BYTES + mat * A_MAT_BYTES + q * 1024 + row * 16);
      async_b128(lds, gsrc);
    }
  };

  const v8f vzero = {0.f, 0.f, 0.f, 0.f, 0.f, 0.f, 0.f, 0.f};
  v8f acc[8];
#pragma unroll
  for (int t = 0; t < 8; ++t) acc[t] = vzero;

  union Frag { v16bf v; u32x4 q[2]; };

  stage(0, 0);
  wait_async0();
  __syncthreads();

  const int rowL = mw * 16 + l15;           // A row within the 64-row block

  for (int kt = 0; kt < D_DIM / 32; ++kt) {
    const int buf = kt & 1;
    if (kt + 1 < D_DIM / 32) stage(kt + 1, buf ^ 1);

    const unsigned char* bs = &slab[buf][0];
    // A fragment (16-bit 16x32): q[0] <- K k0+16g.., q[1] <- K k0+16g+8..
    Frag ah, al;
    ah.q[0] = *(const u32x4*)(bs + W_ALL_BYTES + (2 * g)     * 1024 + rowL * 16);
    ah.q[1] = *(const u32x4*)(bs + W_ALL_BYTES + (2 * g + 1) * 1024 + rowL * 16);
    al.q[0] = *(const u32x4*)(bs + W_ALL_BYTES + A_MAT_BYTES + (2 * g)     * 1024 + rowL * 16);
    al.q[1] = *(const u32x4*)(bs + W_ALL_BYTES + A_MAT_BYTES + (2 * g + 1) * 1024 + rowL * 16);
#pragma unroll
    for (int t = 0; t < 8; ++t) {
      const int colIdx = colBase + t * 16 + l15;
      // B fragment (32x16): lane 16g+n holds K = k0+16g..+15 of col n -> q-planes 2g,2g+1
      Frag bh, bl;
      bh.q[0] = *(const u32x4*)(bs + (2 * g)     * 16384 + colIdx * 16);
      bh.q[1] = *(const u32x4*)(bs + (2 * g + 1) * 16384 + colIdx * 16);
      bl.q[0] = *(const u32x4*)(bs + W_MAT_BYTES + (2 * g)     * 16384 + colIdx * 16);
      bl.q[1] = *(const u32x4*)(bs + W_MAT_BYTES + (2 * g + 1) * 16384 + colIdx * 16);
      acc[t] = __builtin_amdgcn_wmma_f32_16x16x32_bf16(false, ah.v, false, bh.v,
                                                       (short)0, acc[t], false, false);
      acc[t] = __builtin_amdgcn_wmma_f32_16x16x32_bf16(false, ah.v, false, bl.v,
                                                       (short)0, acc[t], false, false);
      acc[t] = __builtin_amdgcn_wmma_f32_16x16x32_bf16(false, al.v, false, bh.v,
                                                       (short)0, acc[t], false, false);
    }
    wait_async0();
    __syncthreads();
  }

  // Bias add (per column)
#pragma unroll
  for (int t = 0; t < 8; ++t) {
    const float bv = bias[colBase + t * 16 + l15];
#pragma unroll
    for (int j = 0; j < 8; ++j) acc[t][j] += bv;
  }

  // LayerNorm scratch aliased into slab (all slab reads are behind the final barrier).
  float* lsum = (float*)(&slab[0][0]);          // [64][8]
  float* lsq  = lsum + 64 * 8;                  // [64][8]

  // Per-row partial sums over this wave's 128 columns.
  // C/D layout: VGPR j of lane -> row = 8*g + j, col = l15 (+tile base).
#pragma unroll
  for (int j = 0; j < 8; ++j) {
    float s = 0.f, s2 = 0.f;
#pragma unroll
    for (int t = 0; t < 8; ++t) { const float y = acc[t][j]; s += y; s2 += y * y; }
#pragma unroll
    for (int m = 1; m < 16; m <<= 1) {           // 16-lane butterfly (stays in group)
      s  += __shfl_xor(s,  m, 32);
      s2 += __shfl_xor(s2, m, 32);
    }
    if (l15 == 0) {
      const int rowl = mw * 16 + 8 * g + j;
      lsum[rowl * 8 + cw] = s;
      lsq[rowl * 8 + cw]  = s2;
    }
  }
  __syncthreads();

  float mean[8], rstd[8];
#pragma unroll
  for (int j = 0; j < 8; ++j) {
    const int rowl = mw * 16 + 8 * g + j;
    float s = 0.f, s2 = 0.f;
#pragma unroll
    for (int c = 0; c < 8; ++c) { s += lsum[rowl * 8 + c]; s2 += lsq[rowl * 8 + c]; }
    const float mu  = s * (1.f / D_DIM);
    const float var = s2 * (1.f / D_DIM) - mu * mu;
    mean[j] = mu;
    rstd[j] = rsqrtf(var + 1e-5f);
  }

#pragma unroll
  for (int t = 0; t < 8; ++t) {
    const int col = colBase + t * 16 + l15;
    const float ga = gamma[col], be = beta[col];
#pragma unroll
    for (int j = 0; j < 8; ++j) {
      const int rowGlob = rowBase + 8 * g + j;
      Out[(size_t)rowGlob * D_DIM + col] = (acc[t][j] - mean[j]) * rstd[j] * ga + be;
    }
  }
}

// ---------------- Launch ----------------------------------------------------------------
extern "C" void kernel_launch(void* const* d_in, const int* in_sizes, int n_in,
                              void* d_out, int out_size, void* d_ws, size_t ws_size,
                              hipStream_t stream) {
  (void)in_sizes; (void)n_in; (void)out_size; (void)ws_size;
  const float* Z     = (const float*)d_in[0];
  const float* decay = (const float*)d_in[1];
  const float* W     = (const float*)d_in[2];
  const float* bias  = (const float*)d_in[3];
  const float* gamma = (const float*)d_in[4];
  const float* beta  = (const float*)d_in[5];
  float* Out = (float*)d_out;

  char* ws = (char*)d_ws;
  size_t off = 0;
  unsigned short* Shi = (unsigned short*)(ws + off); off += (size_t)BN * D_DIM * 2;      // 64 MB
  unsigned short* Slo = (unsigned short*)(ws + off); off += (size_t)BN * D_DIM * 2;      // 64 MB
  unsigned short* Whi = (unsigned short*)(ws + off); off += (size_t)D_DIM * D_DIM * 2;   //  2 MB
  unsigned short* Wlo = (unsigned short*)(ws + off); off += (size_t)D_DIM * D_DIM * 2;   //  2 MB
  float* E = (float*)(ws + off); off += (size_t)B_DIM * NC * D_DIM * 4;                  //  1 MB
  float* C = (float*)(ws + off); off += (size_t)B_DIM * NC * D_DIM * 4;                  //  1 MB

  ema_local_scan<<<dim3(D_DIM / 256, NC, B_DIM), 256, 0, stream>>>(Z, decay, Shi, Slo, E);
  ema_chunk_carry<<<dim3(D_DIM / 256, B_DIM), 256, 0, stream>>>(decay, E, C);
  ema_fixup<<<dim3(D_DIM / 256, N_DIM, B_DIM), 256, 0, stream>>>(decay, C, Shi, Slo);
  w_split<<<dim3((D_DIM * D_DIM) / 256), 256, 0, stream>>>(W, Whi, Wlo);
  gemm_ln<<<dim3(BN / 64), 1024, 0, stream>>>(Shi, Slo, Whi, Wlo, bias, gamma, beta, Out);
}
